// Block_14431090114723
// MI455X (gfx1250) — matrix operand order repaired and checked by reading, not compiled, not run on hardware
//
#include <hip/hip_runtime.h>
#include <hip/hip_bf16.h>
#include <math.h>

// ---------------------------------------------------------------------------
// Transformer block on MI455X (gfx1250, wave32, WMMA).
// Compute-bound (~412 GFLOP vs ~300MB @ 23.3TB/s) -> all matmuls on
// v_wmma_f32_16x16x32_bf16 (fp32 accum). GEMMs: 128x64x64 tiles, 8 WMMA per
// wave per k-tile, double-buffered LDS, async global->LDS staging via the
// gfx1250 GLOBAL_LOAD_ASYNC_TO_LDS_B128 instruction (inline asm; the clang
// builtin has a descriptor-based signature on this toolchain).
// ---------------------------------------------------------------------------

typedef __bf16 v16bf __attribute__((ext_vector_type(16)));
typedef float  v8f   __attribute__((ext_vector_type(8)));

constexpr int B_  = 4;
constexpr int S_  = 2048;
constexpr int E_  = 1024;
constexpr int H_  = 16;
constexpr int DH_ = 64;
constexpr int M_  = B_ * S_;          // 8192 flattened tokens

// 16-byte global->LDS copy; async (ASYNCcnt-tracked) on gfx1250.
__device__ inline void copy16_g2l(const __bf16* __restrict__ g, __bf16* __restrict__ l) {
#if defined(__gfx1250__)
    // LDS address is taken from the VGPR; low 32 bits of the generic pointer are the
    // LDS byte offset (flat->LDS mapping truncates to addr[31:0]).
    unsigned lds_off = (unsigned)(unsigned long long)(uintptr_t)l;
    asm volatile("global_load_async_to_lds_b128 %0, %1, off"
                 :: "v"(lds_off), "v"(g)
                 : "memory");
#else
    *(uint4*)(void*)l = *(const uint4*)(const void*)g;
#endif
}

__device__ inline void async_join() {
#if defined(__gfx1250__)
    asm volatile("s_wait_asynccnt 0" ::: "memory");
#endif
}

// ----------------------------- fragments -----------------------------------

// A 16x32 bf16 fragment (ISA 7.12.2): lane holds row M=lane&15; lane half
// selects K base 0/8, upper vgprs at +16. 16B-contiguous -> ds_load_b128.
__device__ inline v16bf ldfrag_A(const __bf16* __restrict__ base, int stride, int lane) {
    int m  = lane & 15;
    int kb = (lane >> 4) * 8;
    const __bf16* r = base + m * stride;
    v16bf a;
#pragma unroll
    for (int j = 0; j < 8; ++j) a[j]     = r[kb + j];
#pragma unroll
    for (int j = 0; j < 8; ++j) a[8 + j] = r[16 + kb + j];
    return a;
}

// B 32x16 bf16 fragment from a TRANSPOSED [N][K] LDS tile: lane holds col
// N=lane&15; lane half selects K base 0/8, elements 8..15 at +16 (matches A).
__device__ inline v16bf ldfrag_B(const __bf16* __restrict__ baseT, int stride, int lane) {
    int n  = lane & 15;
    int kb = (lane >> 4) * 8;
    const __bf16* r = baseT + n * stride;
    v16bf b;
#pragma unroll
    for (int j = 0; j < 16; ++j) b[j] = r[kb + j + (j >= 8 ? 8 : 0)];
    return b;
}

#define WMMA_BF16(A, Bf, C) \
    __builtin_amdgcn_wmma_f32_16x16x32_bf16(false, (A), false, (Bf), (short)0, (C), false, false)

// ----------------------- fp32 -> bf16 conversion ---------------------------

__global__ __launch_bounds__(256)
void cvt_bf16_kernel(const float* __restrict__ in, __bf16* __restrict__ out, int n) {
    int i = blockIdx.x * 256 + threadIdx.x;
    if (i < n) out[i] = (__bf16)in[i];
}

// ------------------------------ layernorm ----------------------------------

__global__ __launch_bounds__(256)
void layernorm_bf16_kernel(const float* __restrict__ x, const float* __restrict__ g,
                           const float* __restrict__ bta, __bf16* __restrict__ out) {
    __shared__ float redS[8];
    __shared__ float redQ[8];
    int row = blockIdx.x;
    const float* xr = x + (size_t)row * E_;
    float v[4];
    float s = 0.f, ss = 0.f;
#pragma unroll
    for (int i = 0; i < 4; ++i) {
        v[i] = xr[threadIdx.x + i * 256];
        s += v[i];
        ss += v[i] * v[i];
    }
#pragma unroll
    for (int m = 1; m < 32; m <<= 1) {
        s  += __shfl_xor(s,  m, 32);
        ss += __shfl_xor(ss, m, 32);
    }
    int wave = threadIdx.x >> 5, lane = threadIdx.x & 31;
    if (lane == 0) { redS[wave] = s; redQ[wave] = ss; }
    __syncthreads();
    float ts = 0.f, tq = 0.f;
#pragma unroll
    for (int i = 0; i < 8; ++i) { ts += redS[i]; tq += redQ[i]; }
    float mu   = ts * (1.0f / E_);
    float var  = tq * (1.0f / E_) - mu * mu;
    float rstd = rsqrtf(var + 1e-5f);
#pragma unroll
    for (int i = 0; i < 4; ++i) {
        int c = threadIdx.x + i * 256;
        out[(size_t)row * E_ + c] = (__bf16)((v[i] - mu) * rstd * g[c] + bta[c]);
    }
}

// ------------------------------- GEMM --------------------------------------
// C[M,N] = A[M,K](bf16) * W[K,N](bf16) (+bias)(+residual)(ReLU) -> f32/bf16.
// Tile 128x64x64, 256 thr = 8 waves, wave owns 2x2 fragments -> 8 WMMA/tile.
// Double-buffered LDS; A staged async, B (transposed) loads issued before
// compute, scattered after.

template<bool HEADED_W, bool HEADED_OUT, bool OUT_BF16, bool HAS_BIAS, bool HAS_RES, bool RELU>
__global__ __launch_bounds__(256)
void gemm_bf16_kernel(const __bf16* __restrict__ A, const __bf16* __restrict__ W,
                      const float* __restrict__ bias, const float* __restrict__ resid,
                      void* __restrict__ outp, int M, int N, int K) {
    constexpr int BM = 128, BN = 64, BK = 64, LDT = BK + 8;  // 144B rows: 16B aligned
    __shared__ __bf16 sA[2][BM][LDT];
    __shared__ __bf16 sB[2][BN][LDT];   // transposed [n][k]

    int m0 = blockIdx.x * BM, n0 = blockIdx.y * BN;
    int tid = threadIdx.x, wave = tid >> 5, lane = tid & 31;
    int wm = (wave >> 1) * 2;   // M frags {wm, wm+1}
    int wn = (wave & 1) * 2;    // N frags {wn, wn+1}

    // staging coordinates
    int arow = tid >> 1;              // 0..127
    int acol = (tid & 1) * 32;
    int bk   = tid >> 2;              // 0..63
    int bn   = (tid & 3) * 16;

    auto stageA = [&](int kt, int buf) {
        const __bf16* g = A + (size_t)(m0 + arow) * K + kt * BK + acol;
        __bf16* l = &sA[buf][arow][acol];
#pragma unroll
        for (int j = 0; j < 4; ++j) copy16_g2l(g + j * 8, l + j * 8);
    };
    auto loadB = [&](int kt, uint4& d0, uint4& d1) {
        const __bf16* g;
        if (HEADED_W) {
            int c = n0 + bn;
            g = W + ((size_t)(c >> 6) * K + (kt * BK + bk)) * 64 + (c & 63);
        } else {
            g = W + (size_t)(kt * BK + bk) * N + n0 + bn;
        }
        d0 = *(const uint4*)(const void*)g;
        d1 = *(const uint4*)(const void*)(g + 8);
    };
    auto scatterB = [&](int buf, const uint4& d0, const uint4& d1) {
        const __bf16* e0 = (const __bf16*)&d0;
        const __bf16* e1 = (const __bf16*)&d1;
#pragma unroll
        for (int j = 0; j < 8; ++j) sB[buf][bn + j][bk]     = e0[j];
#pragma unroll
        for (int j = 0; j < 8; ++j) sB[buf][bn + 8 + j][bk] = e1[j];
    };

    v8f acc00 = {}, acc01 = {}, acc10 = {}, acc11 = {};

    int nk = K / BK;
    // prologue: stage tile 0 into buffer 0
    stageA(0, 0);
    {
        uint4 d0, d1;
        loadB(0, d0, d1);
        scatterB(0, d0, d1);
    }

    for (int kt = 0; kt < nk; ++kt) {
        int cur = kt & 1, nxt = cur ^ 1;
        async_join();
        __syncthreads();                        // tile kt resident in LDS

        bool havenext = (kt + 1) < nk;
        uint4 d0, d1;
        if (havenext) {
            stageA(kt + 1, nxt);                // async fill of next A tile
            loadB(kt + 1, d0, d1);              // issue next B loads (no wait yet)
        }

#pragma unroll
        for (int ks = 0; ks < 2; ++ks) {
            int ko = ks * 32;
            v16bf a0 = ldfrag_A(&sA[cur][(wm    ) * 16][ko], LDT, lane);
            v16bf a1 = ldfrag_A(&sA[cur][(wm + 1) * 16][ko], LDT, lane);
            v16bf b0 = ldfrag_B(&sB[cur][(wn    ) * 16][ko], LDT, lane);
            v16bf b1 = ldfrag_B(&sB[cur][(wn + 1) * 16][ko], LDT, lane);
            acc00 = WMMA_BF16(a0, b0, acc00);
            acc01 = WMMA_BF16(a0, b1, acc01);
            acc10 = WMMA_BF16(a1, b0, acc10);
            acc11 = WMMA_BF16(a1, b1, acc11);
        }

        if (havenext) scatterB(nxt, d0, d1);    // transpose-store after compute
    }

    // epilogue — C layout: vgpr r -> row r + 8*(lane>=16), col = lane&15
    int col = lane & 15;
    int rb  = (lane >> 4) * 8;
#pragma unroll
    for (int im = 0; im < 2; ++im) {
#pragma unroll
        for (int in = 0; in < 2; ++in) {
            v8f& acc = (im == 0) ? ((in == 0) ? acc00 : acc01)
                                 : ((in == 0) ? acc10 : acc11);
            int n = n0 + (wn + in) * 16 + col;
            float bval = HAS_BIAS ? bias[n] : 0.f;
#pragma unroll
            for (int r = 0; r < 8; ++r) {
                int m = m0 + (wm + im) * 16 + rb + r;
                float val = acc[r] + bval;
                if (RELU)    val = fmaxf(val, 0.f);
                if (HAS_RES) val += resid[(size_t)m * N + n];
                size_t oi;
                if (HEADED_OUT) {
                    int bb = m / S_, ssr = m % S_;
                    int hh = n >> 6, dd = n & 63;
                    oi = (((size_t)bb * H_ + hh) * S_ + ssr) * DH_ + dd;
                } else {
                    oi = (size_t)m * N + n;
                }
                if (OUT_BF16) ((__bf16*)outp)[oi] = (__bf16)val;
                else          ((float*)outp)[oi]  = val;
            }
        }
    }
}

// --------------------------- flash attention -------------------------------
// Grid (S/64, H, B), 128 thr = 4 waves; wave owns 16 query rows. Causal,
// fp32 online softmax; 16 WMMA per wave per 64-key tile.

__global__ __launch_bounds__(128)
void attn_kernel(const __bf16* __restrict__ q, const __bf16* __restrict__ k,
                 const __bf16* __restrict__ v, __bf16* __restrict__ o) {
    __shared__ __bf16 sK[64][64];          // [t][d]  (row stride 128B)
    __shared__ __bf16 sVT[64][64 + 8];     // [d][t]  (row stride 144B)
    __shared__ __bf16 sP[4][16][64 + 8];   // per-wave P tile [m][t]

    int qblk = blockIdx.x;
    int h = blockIdx.y, b = blockIdx.z;
    int tid = threadIdx.x, wave = tid >> 5, lane = tid & 31;
    const size_t headoff = (((size_t)b * H_) + h) * S_ * DH_;
    int s0 = qblk * 64 + wave * 16;
    int col16 = lane & 15;
    int rb    = (lane >> 4) * 8;

    // Q fragments (16x64) straight from global into A-layout registers
    v16bf qf[2];
    {
        int m  = lane & 15;
        int kb = (lane >> 4) * 8;
        const __bf16* qr = q + headoff + (size_t)(s0 + m) * DH_;
#pragma unroll
        for (int kk = 0; kk < 2; ++kk) {
            v16bf a;
#pragma unroll
            for (int j = 0; j < 8; ++j) a[j]     = qr[kk * 32 + kb + j];
#pragma unroll
            for (int j = 0; j < 8; ++j) a[8 + j] = qr[kk * 32 + 16 + kb + j];
            qf[kk] = a;
        }
    }

    float m_i[8], l_i[8];
#pragma unroll
    for (int r = 0; r < 8; ++r) { m_i[r] = -1e30f; l_i[r] = 0.f; }
    v8f oacc[4];
#pragma unroll
    for (int f = 0; f < 4; ++f) { v8f z = {}; oacc[f] = z; }

    int ntiles = qblk + 1;                     // causal
    for (int t = 0; t < ntiles; ++t) {
        int t0 = t * 64;
        {   // K tile [t][d] — async contiguous staging
            int row = tid >> 1;
            int c   = (tid & 1) * 32;
            const __bf16* g = k + headoff + (size_t)(t0 + row) * DH_ + c;
#pragma unroll
            for (int j = 0; j < 4; ++j) copy16_g2l(g + j * 8, &sK[row][c + j * 8]);
        }
        {   // V tile transposed -> [d][t]
            int row = tid >> 1;
            int c   = (tid & 1) * 32;
            const __bf16* g = v + headoff + (size_t)(t0 + row) * DH_ + c;
            uint4 d0 = *(const uint4*)(const void*)g;
            uint4 d1 = *(const uint4*)(const void*)(g + 8);
            uint4 d2 = *(const uint4*)(const void*)(g + 16);
            uint4 d3 = *(const uint4*)(const void*)(g + 24);
            const __bf16* e = (const __bf16*)&d0;
#pragma unroll
            for (int j = 0; j < 8; ++j) sVT[c + j][row] = e[j];
            e = (const __bf16*)&d1;
#pragma unroll
            for (int j = 0; j < 8; ++j) sVT[c + 8 + j][row] = e[j];
            e = (const __bf16*)&d2;
#pragma unroll
            for (int j = 0; j < 8; ++j) sVT[c + 16 + j][row] = e[j];
            e = (const __bf16*)&d3;
#pragma unroll
            for (int j = 0; j < 8; ++j) sVT[c + 24 + j][row] = e[j];
        }
        async_join();
        __syncthreads();

        // S = Q K^T : 4 N-fragments, k-dim = DH (2 steps of 32)
        v8f sc[4];
#pragma unroll
        for (int f = 0; f < 4; ++f) {
            v8f a = {};
            int n  = f * 16 + col16;
            int kb = (lane >> 4) * 8;
#pragma unroll
            for (int kk = 0; kk < 2; ++kk) {
                v16bf bf;
#pragma unroll
                for (int j = 0; j < 16; ++j) bf[j] = sK[n][kk * 32 + kb + j + (j >= 8 ? 8 : 0)];
                a = WMMA_BF16(qf[kk], bf, a);
            }
            sc[f] = a;
        }

        // scale + causal mask (arithmetic; EXEC stays full for WMMA)
#pragma unroll
        for (int f = 0; f < 4; ++f) {
            int tcol = t0 + f * 16 + col16;
#pragma unroll
            for (int r = 0; r < 8; ++r) {
                int srow = s0 + rb + r;
                float xv = sc[f][r] * 0.125f;   // 1/sqrt(64)
                sc[f][r] = (tcol <= srow) ? xv : -1e30f;
            }
        }

        // online softmax (row stats replicated across the 16 lanes of a row)
        float pLoc[4][8];
#pragma unroll
        for (int r = 0; r < 8; ++r) {
            float tmax = fmaxf(fmaxf(sc[0][r], sc[1][r]), fmaxf(sc[2][r], sc[3][r]));
#pragma unroll
            for (int msk = 1; msk < 16; msk <<= 1) tmax = fmaxf(tmax, __shfl_xor(tmax, msk, 32));
            float mnew  = fmaxf(m_i[r], tmax);
            float alpha = __expf(m_i[r] - mnew);
            m_i[r] = mnew;
            l_i[r] *= alpha;
#pragma unroll
            for (int f = 0; f < 4; ++f) oacc[f][r] *= alpha;
            float psum = 0.f;
#pragma unroll
            for (int f = 0; f < 4; ++f) {
                float p = __expf(sc[f][r] - mnew);
                pLoc[f][r] = p;
                psum += p;
            }
#pragma unroll
            for (int msk = 1; msk < 16; msk <<= 1) psum += __shfl_xor(psum, msk, 32);
            l_i[r] += psum;
        }

        // C-layout -> LDS [m][t] so P can be reloaded in A-layout
#pragma unroll
        for (int f = 0; f < 4; ++f)
#pragma unroll
            for (int r = 0; r < 8; ++r)
                sP[wave][rb + r][f * 16 + col16] = (__bf16)pLoc[f][r];
        __syncthreads();

        v16bf pf[2];
        {
            int m  = lane & 15;
            int kb = (lane >> 4) * 8;
#pragma unroll
            for (int kk = 0; kk < 2; ++kk) {
                v16bf a;
#pragma unroll
                for (int j = 0; j < 8; ++j) a[j]     = sP[wave][m][kk * 32 + kb + j];
#pragma unroll
                for (int j = 0; j < 8; ++j) a[8 + j] = sP[wave][m][kk * 32 + 16 + kb + j];
                pf[kk] = a;
            }
        }

        // O += P * V (4 d-fragments, k-dim = t, 2 steps of 32)
#pragma unroll
        for (int f = 0; f < 4; ++f) {
            int n  = f * 16 + col16;
            int kb = (lane >> 4) * 8;
#pragma unroll
            for (int kk = 0; kk < 2; ++kk) {
                v16bf bf;
#pragma unroll
                for (int j = 0; j < 16; ++j) bf[j] = sVT[n][kk * 32 + kb + j + (j >= 8 ? 8 : 0)];
                oacc[f] = WMMA_BF16(pf[kk], bf, oacc[f]);
            }
        }
        __syncthreads();   // protect sK/sVT/sP before next tile overwrite
    }

    // normalize + scatter heads into [B,S,E] bf16 (input of out-proj GEMM)
#pragma unroll
    for (int f = 0; f < 4; ++f) {
#pragma unroll
        for (int r = 0; r < 8; ++r) {
            int srow = s0 + rb + r;
            float val = oacc[f][r] / l_i[r];
            o[((size_t)b * S_ + srow) * E_ + h * DH_ + f * 16 + col16] = (__bf16)val;
        }
    }
}

// ------------------------------ launcher -----------------------------------

extern "C" void kernel_launch(void* const* d_in, const int* in_sizes, int n_in,
                              void* d_out, int out_size, void* d_ws, size_t ws_size,
                              hipStream_t stream) {
    const float* x     = (const float*)d_in[0];
    const float* Wq    = (const float*)d_in[1];
    const float* Wk    = (const float*)d_in[2];
    const float* Wv    = (const float*)d_in[3];
    const float* Wo    = (const float*)d_in[4];
    const float* bo    = (const float*)d_in[5];
    const float* ln1_g = (const float*)d_in[6];
    const float* ln1_b = (const float*)d_in[7];
    const float* ln2_g = (const float*)d_in[8];
    const float* ln2_b = (const float*)d_in[9];
    const float* W1    = (const float*)d_in[10];
    const float* b1    = (const float*)d_in[11];
    const float* W2    = (const float*)d_in[12];
    const float* b2    = (const float*)d_in[13];
    float* out = (float*)d_out;

    // workspace layout (256B-aligned slabs)
    char* w = (char*)d_ws;
    auto take = [&](size_t bytes) { char* p = w; w += (bytes + 255) & ~size_t(255); return p; };
    __bf16* lx1 = (__bf16*)take((size_t)M_ * E_ * 2);
    __bf16* qb  = (__bf16*)take((size_t)M_ * E_ * 2);
    __bf16* kb  = (__bf16*)take((size_t)M_ * E_ * 2);
    __bf16* vb  = (__bf16*)take((size_t)M_ * E_ * 2);
    __bf16* ob  = (__bf16*)take((size_t)M_ * E_ * 2);
    float*  x2  = (float*) take((size_t)M_ * E_ * 4);
    __bf16* lx2 = (__bf16*)take((size_t)M_ * E_ * 2);
    __bf16* h1  = (__bf16*)take((size_t)M_ * 4 * E_ * 2);
    __bf16* wqb = (__bf16*)take((size_t)H_ * E_ * DH_ * 2);
    __bf16* wkb = (__bf16*)take((size_t)H_ * E_ * DH_ * 2);
    __bf16* wvb = (__bf16*)take((size_t)H_ * E_ * DH_ * 2);
    __bf16* wob = (__bf16*)take((size_t)E_ * E_ * 2);
    __bf16* w1b = (__bf16*)take((size_t)E_ * 4 * E_ * 2);
    __bf16* w2b = (__bf16*)take((size_t)4 * E_ * E_ * 2);

    auto cvt = [&](const float* src, __bf16* dst, int n) {
        cvt_bf16_kernel<<<(n + 255) / 256, 256, 0, stream>>>(src, dst, n);
    };
    cvt(Wq, wqb, H_ * E_ * DH_);
    cvt(Wk, wkb, H_ * E_ * DH_);
    cvt(Wv, wvb, H_ * E_ * DH_);
    cvt(Wo, wob, E_ * E_);
    cvt(W1, w1b, E_ * 4 * E_);
    cvt(W2, w2b, 4 * E_ * E_);

    // LN1 -> bf16
    layernorm_bf16_kernel<<<M_, 256, 0, stream>>>(x, ln1_g, ln1_b, lx1);

    // QKV projections (headed weights, [B,H,S,DH] bf16 outputs)
    dim3 gq(M_ / 128, E_ / 64);
    gemm_bf16_kernel<true, true, true, false, false, false><<<gq, 256, 0, stream>>>(
        lx1, wqb, nullptr, nullptr, qb, M_, E_, E_);
    gemm_bf16_kernel<true, true, true, false, false, false><<<gq, 256, 0, stream>>>(
        lx1, wkb, nullptr, nullptr, kb, M_, E_, E_);
    gemm_bf16_kernel<true, true, true, false, false, false><<<gq, 256, 0, stream>>>(
        lx1, wvb, nullptr, nullptr, vb, M_, E_, E_);

    // causal flash attention -> concat heads [B,S,E] bf16
    attn_kernel<<<dim3(S_ / 64, H_, B_), 128, 0, stream>>>(qb, kb, vb, ob);

    // out-proj + bias + residual -> x2 (fp32 residual stream)
    gemm_bf16_kernel<false, false, false, true, true, false><<<gq, 256, 0, stream>>>(
        ob, wob, bo, x, x2, M_, E_, E_);

    // LN2 -> bf16
    layernorm_bf16_kernel<<<M_, 256, 0, stream>>>(x2, ln2_g, ln2_b, lx2);

    // FFN1: relu(lx2 @ W1 + b1) -> bf16
    gemm_bf16_kernel<false, false, true, true, false, true><<<dim3(M_ / 128, 4 * E_ / 64), 256, 0, stream>>>(
        lx2, w1b, b1, nullptr, h1, M_, 4 * E_, E_);

    // FFN2: x2 + h1 @ W2 + b2 -> fp32 d_out
    gemm_bf16_kernel<false, false, false, true, true, false><<<gq, 256, 0, stream>>>(
        h1, w2b, b2, x2, out, M_, E_, 4 * E_);
}
